// Quantizer_1279900254416
// MI455X (gfx1250) — matrix operand order repaired and checked by reading, not compiled, not run on hardware
//
#include <hip/hip_runtime.h>
#include <hip/hip_bf16.h>

// VQ-VAE quantizer for MI455X (gfx1250, wave32, WMMA).
// distances argmin reduced to argmax(2*f.w_k - ||w_k||^2)  (||f||^2 is row-constant).
// Matmul f[N,64] @ w[64,512] done with v_wmma_f32_16x16x32_f16.

typedef __attribute__((ext_vector_type(16))) _Float16 v16h;
typedef __attribute__((ext_vector_type(8)))  float    v8f;
typedef __attribute__((ext_vector_type(2)))  _Float16 h2;

#define D_DIM   64
#define K_DIM   512
#define KHALF   256     // codes per LDS phase
#define WTS     68      // LDS row stride in halves (padded: 64 + 4) to dodge bank conflicts
#define ROWS_PER_BLOCK 128
#define NEG_INF (-3.402823466e38f)

__global__ void vq_init_loss(float* loss) {
    if (threadIdx.x == 0 && blockIdx.x == 0) *loss = 0.0f;
}

__global__ __launch_bounds__(256)
void vq_quantize_kernel(const float* __restrict__ x,      // [N, 64]
                        const float* __restrict__ w,      // [64, 512] row-major (d-major)
                        int*   __restrict__ out_idx,      // [N]
                        float* __restrict__ out_zq,       // [N, 64]
                        float* __restrict__ out_ze,       // [N, 64]
                        float* __restrict__ out_loss,     // [1]
                        float  loss_scale) {
    __shared__ _Float16 wT[KHALF * WTS];   // wT[k_local][d] as f16, padded stride
    __shared__ float    wsq[KHALF];        // ||w_k||^2 for current phase
    __shared__ float    lsum;

    const int tid      = threadIdx.x;
    const int lane     = tid & 31;
    const int wave     = tid >> 5;             // 0..7
    const int colc     = lane & 15;            // column within 16-wide tile / C-layout N
    const int half_sel = lane >> 4;            // 0 or 1 (lane group)
    const int m0       = blockIdx.x * ROWS_PER_BLOCK + wave * 16;

    // ---------------- Load A fragments (this lane's x row) + emit z_e ----------------
    // A 16x32 f16 layout: lane(0..15)=row M, koff(v) = (v<4?0:16) + 8*half_sel + 2*(v&3).
    // Union of K indices needed by this lane over both k-chunks (kc=0,32):
    //   [b8,b8+8) u [16+b8,+8) u [32+b8,+8) u [48+b8,+8),  b8 = 8*half_sel.
    const int   rowm = m0 + colc;
    const int   b8   = half_sel * 8;
    const float* xr  = x + (size_t)rowm * D_DIM + b8;

    float4 q0 = *(const float4*)(xr +  0);
    float4 q1 = *(const float4*)(xr +  4);
    float4 q2 = *(const float4*)(xr + 16);
    float4 q3 = *(const float4*)(xr + 20);
    float4 q4 = *(const float4*)(xr + 32);
    float4 q5 = *(const float4*)(xr + 36);
    float4 q6 = *(const float4*)(xr + 48);
    float4 q7 = *(const float4*)(xr + 52);

    // z_e = x : write back the same data (lanes L and L+16 cover complementary halves)
    float* zer = out_ze + (size_t)rowm * D_DIM + b8;
    *(float4*)(zer +  0) = q0;  *(float4*)(zer +  4) = q1;
    *(float4*)(zer + 16) = q2;  *(float4*)(zer + 20) = q3;
    *(float4*)(zer + 32) = q4;  *(float4*)(zer + 36) = q5;
    *(float4*)(zer + 48) = q6;  *(float4*)(zer + 52) = q7;

    v16h A0, A1;
#define PUT4(dst, base, q)                         \
    dst[base + 0] = (_Float16)(q).x;               \
    dst[base + 1] = (_Float16)(q).y;               \
    dst[base + 2] = (_Float16)(q).z;               \
    dst[base + 3] = (_Float16)(q).w;
    PUT4(A0,  0, q0) PUT4(A0,  4, q1) PUT4(A0,  8, q2) PUT4(A0, 12, q3)
    PUT4(A1,  0, q4) PUT4(A1,  4, q5) PUT4(A1,  8, q6) PUT4(A1, 12, q7)
#undef PUT4

    float bestS[8];
    int   bestI[8];
#pragma unroll
    for (int r = 0; r < 8; ++r) { bestS[r] = NEG_INF; bestI[r] = 0; }

    if (tid == 0) lsum = 0.0f;

    // ---------------- Two phases over the codebook (256 codes each) ----------------
    for (int p = 0; p < 2; ++p) {
        __syncthreads();   // protect wT against readers from previous phase

        // Stage w[:, p*256 .. p*256+255] transposed into LDS as f16.
        for (int i = tid; i < D_DIM * KHALF; i += 256) {
            int d  = i >> 8;      // 0..63
            int kl = i & 255;     // 0..255
            wT[kl * WTS + d] = (_Float16)w[(size_t)d * K_DIM + p * KHALF + kl];
        }

        // While phase 0 computes, pull the phase-1 half of the codebook toward the
        // WGP: 64 rows x 1 KB = 64 KB = 512 cachelines -> global_prefetch_b8.
        if (p == 0) {
#pragma unroll
            for (int i = tid; i < 64 * 8; i += 256) {
                int d = i >> 3;       // row (dim)
                int c = i & 7;        // 128B cacheline within the phase-1 span
                __builtin_prefetch(w + (size_t)d * K_DIM + KHALF + c * 32, 0, 3);
            }
        }
        __syncthreads();

        // ||w_k||^2 from the (f16-rounded) staged codes — consistent with WMMA dots.
        {
            int kl = tid;  // blockDim == 256 == KHALF
            float s = 0.0f;
            const _Float16* wr = wT + kl * WTS;
#pragma unroll
            for (int d = 0; d < D_DIM; d += 2) {
                h2 v = *(const h2*)(wr + d);
                float a = (float)v[0], b = (float)v[1];
                s += a * a + b * b;
            }
            wsq[kl] = s;
        }
        __syncthreads();

        // 16 column-tiles of 16 codes each; 2 WMMAs (k-chunks) per tile.
        // Unroll x2 so tile t+1's LDS loads + WMMAs overlap tile t's score update
        // (kills the 4-NOP WMMA->VALU RAW bubble; WMMA co-executes as TRANS).
#pragma unroll 2
        for (int t = 0; t < 16; ++t) {
            const int kl = t * 16 + colc;                       // local column 0..255
            const _Float16* bp = wT + kl * WTS + half_sel * 8;  // B layout: K = 8*hs + 2v (+8 for v>=4)
            v16h B0, B1;
#pragma unroll
            for (int v = 0; v < 8; ++v) {
                h2 p0 = *(const h2*)(bp + 2 * v + ((v & 4) ? 8 : 0));        // kc = 0
                h2 p1 = *(const h2*)(bp + 32 + 2 * v + ((v & 4) ? 8 : 0));   // kc = 32
                B0[2 * v] = p0[0]; B0[2 * v + 1] = p0[1];
                B1[2 * v] = p1[0]; B1[2 * v + 1] = p1[1];
            }

            v8f c = {};
            c = __builtin_amdgcn_wmma_f32_16x16x32_f16(false, A0, false, B0,
                                                       (short)0, c, false, false);
            c = __builtin_amdgcn_wmma_f32_16x16x32_f16(false, A1, false, B1,
                                                       (short)0, c, false, false);

            const float ws = wsq[kl];
            const int   gk = p * KHALF + kl;
#pragma unroll
            for (int r = 0; r < 8; ++r) {
                float s = 2.0f * c[r] - ws;      // rank-equivalent to -distance
                if (s > bestS[r]) { bestS[r] = s; bestI[r] = gk; }  // '>' keeps first idx
            }
        }
    }

    // ---------------- Argmax reduction across the 16 column-lanes ----------------
#pragma unroll
    for (int r = 0; r < 8; ++r) {
        float s  = bestS[r];
        int   bi = bestI[r];
#pragma unroll
        for (int m = 1; m <= 8; m <<= 1) {
            float os = __shfl_xor(s,  m, 32);
            int   oi = __shfl_xor(bi, m, 32);
            if (os > s || (os == s && oi < bi)) { s = os; bi = oi; }
        }
        bestS[r] = s;
        bestI[r] = bi;   // uniform across the 16-lane group; row = r + 8*half_sel
    }

    if (colc == 0) {
#pragma unroll
        for (int r = 0; r < 8; ++r)
            out_idx[m0 + 8 * half_sel + r] = bestI[r];
    }

    // ---------------- z_q scatter (f32 codebook) + loss accumulation ----------------
    float lacc = 0.0f;
#pragma unroll
    for (int r = 0; r < 8; ++r) {
        const int    m = m0 + 8 * half_sel + r;
        const int    k = bestI[r];
#pragma unroll
        for (int j = 0; j < 4; ++j) {
            int   d  = colc + 16 * j;                     // 16 lanes cover all 64 dims
            float wv = w[(size_t)d * K_DIM + k];
            float xv = x[(size_t)m * D_DIM + d];
            out_zq[(size_t)m * D_DIM + d] = wv;           // z_q_st == z_q numerically
            float df = wv - xv;
            lacc += df * df;
        }
    }

    atomicAdd(&lsum, lacc);
    __syncthreads();
    if (tid == 0) atomicAdd(out_loss, lsum * loss_scale); // scale = 2/(N*D): q + 1.0*e loss
}

extern "C" void kernel_launch(void* const* d_in, const int* in_sizes, int n_in,
                              void* d_out, int out_size, void* d_ws, size_t ws_size,
                              hipStream_t stream) {
    const float* x = (const float*)d_in[0];   // [N, 64] f32
    const float* w = (const float*)d_in[1];   // [64, 512] f32

    const int N = in_sizes[0] / D_DIM;        // 131072

    float* out      = (float*)d_out;
    int*   out_idx  = (int*)d_out;                     // N int32 (bit-stored)
    float* out_zq   = out + N;                         // N*64
    float* out_ze   = out_zq + (size_t)N * D_DIM;      // N*64
    float* out_loss = out_ze + (size_t)N * D_DIM;      // 1

    const float loss_scale = 2.0f / (float)((size_t)N * D_DIM);

    vq_init_loss<<<1, 1, 0, stream>>>(out_loss);

    dim3 grid(N / ROWS_PER_BLOCK);   // 1024 (exact, no tail -> EXEC all-ones for WMMA)
    dim3 block(256);                 // 8 wave32 -> 128 rows per block
    vq_quantize_kernel<<<grid, block, 0, stream>>>(x, w, out_idx, out_zq, out_ze,
                                                   out_loss, loss_scale);
}
